// TreeNetwork_20555713479391
// MI455X (gfx1250) — matrix operand order repaired and checked by reading, not compiled, not run on hardware
//
#include <hip/hip_runtime.h>
#include <hip/hip_bf16.h>

typedef __attribute__((ext_vector_type(16))) _Float16 v16h;
typedef __attribute__((ext_vector_type(8)))  float    v8f;

// ---------------------------------------------------------------------------
// Implicit-GEMM conv (NCHW, OIHW) via WMMA f32 = f16 x f16 + f32.
// Block = 128 threads = 4 waves. All 4 waves share a 16(Cout) x 64(K) weight
// tile staged through LDS; each wave owns a 16-pixel column tile, K unrolled
// by 64 (2 x v_wmma_f32_16x16x32_f16 per iteration).
// A fragment (16x32 f16): lane L holds row M=L%16; half h=L/16;
//   half j (0..15) -> K = kbase + j + 8*h + (j>=8 ? 8 : 0)
// B fragment (32x16 f16): lane L holds col N=L%16; half j -> K = kbase + j + 8*h + (j>=8 ? 8 : 0)
// C/D (16x16 f32): lane L, reg r -> M = r + 8*h, N = L%16
// ---------------------------------------------------------------------------
template<int KS, int PAD>
__global__ __launch_bounds__(128) void conv_wmma_t(
    const float* __restrict__ in, const float* __restrict__ wt,
    float* __restrict__ out, int B, int Cin, int H, int W, int Cout)
{
    constexpr int KK = KS * KS;
    const int HW   = H * W;
    const int b    = blockIdx.z;
    const int co0  = blockIdx.y * 16;
    const int wave = threadIdx.x >> 5;
    const int lane = threadIdx.x & 31;
    const int px0  = blockIdx.x * 64 + wave * 16;
    const int half = lane >> 4;
    const int mn   = lane & 15;
    const int Ktot = Cin * KK;

    const float* inB = in + (size_t)b * Cin * HW;

    __shared__ _Float16 lds_a[16 * 64];   // [m][k] tile of weights, shared by 4 waves

    v8f acc = {};
    const int p   = px0 + mn;
    const int py  = (p < HW) ? (p / W) : 0;
    const int pxx = (p < HW) ? (p % W) : 0;

    for (int k0 = 0; k0 < Ktot; k0 += 64) {
        // cooperative, coalesced weight-tile stage through LDS (f32 -> f16)
        for (int t = threadIdx.x; t < 1024; t += 128) {
            int m  = t >> 6;
            int kk = t & 63;
            int co = co0 + m;
            int kg = k0 + kk;
            float v = (co < Cout && kg < Ktot) ? wt[(size_t)co * Ktot + kg] : 0.0f;
            lds_a[t] = (_Float16)v;
        }
        // prefetch next weight tile (lowers to global_prefetch_b8)
        if (k0 + 64 < Ktot && co0 < Cout)
            __builtin_prefetch(wt + (size_t)co0 * Ktot + k0 + 64, 0, 1);
        __syncthreads();

        #pragma unroll
        for (int s = 0; s < 2; ++s) {
            const int kbase = k0 + 32 * s;
            v16h a, bf;
            #pragma unroll
            for (int j = 0; j < 16; ++j) {
                int kk = 32 * s + j + 8 * half + ((j >= 8) ? 8 : 0);
                a[j] = lds_a[mn * 64 + kk];
            }
            #pragma unroll
            for (int j = 0; j < 16; ++j) {
                int kk = kbase + j + 8 * half + ((j >= 8) ? 8 : 0);
                float v = 0.0f;
                if (p < HW && kk < Ktot) {
                    if constexpr (KS == 1) {
                        v = inB[(size_t)kk * HW + p];
                    } else {
                        int ci = kk / KK;           // compile-time divisor
                        int r  = kk % KK;
                        int y  = py + r / KS - PAD;
                        int x  = pxx + r % KS - PAD;
                        if (y >= 0 && y < H && x >= 0 && x < W)
                            v = inB[(size_t)ci * HW + y * W + x];
                    }
                }
                bf[j] = (_Float16)v;
            }
            acc = __builtin_amdgcn_wmma_f32_16x16x32_f16(
                      false, a, false, bf, (short)0, acc, false, false);
        }
        __syncthreads();
    }

    float* outB = out + (size_t)b * Cout * HW;
    if (p < HW) {
        #pragma unroll
        for (int r = 0; r < 8; ++r) {
            int co = co0 + r + 8 * half;
            if (co < Cout) outB[(size_t)co * HW + p] = acc[r];
        }
    }
}

// ---------------------------------------------------------------------------
// Grid edge distances: sum over 128 channels of squared differences
// ---------------------------------------------------------------------------
__global__ void edge_dists_k(const float* __restrict__ lat,
                             float* __restrict__ wrow, float* __restrict__ wcol,
                             int B, int H, int W)
{
    const int nh = H * (W - 1);
    const int nv = (H - 1) * W;
    const int HW = H * W;
    int i = blockIdx.x * blockDim.x + threadIdx.x;
    if (i >= B * (nh + nv)) return;
    int b = i / (nh + nv);
    int r = i % (nh + nv);
    const float* L = lat + (size_t)b * 128 * HW;
    float s = 0.0f;
    if (r < nh) {
        int row = r / (W - 1), col = r % (W - 1);
        int o = row * W + col;
        for (int c = 0; c < 128; ++c) {
            float d = L[(size_t)c * HW + o] - L[(size_t)c * HW + o + 1];
            s += d * d;
        }
        wrow[(size_t)b * nh + r] = s;
    } else {
        int o = r - nh;   // row*W+col over (H-1)xW
        for (int c = 0; c < 128; ++c) {
            float d = L[(size_t)c * HW + o] - L[(size_t)c * HW + o + W];
            s += d * d;
        }
        wcol[(size_t)b * nv + o] = s;
    }
}

// ---------------------------------------------------------------------------
// MST per image: bitonic stable sort of edges, Kruskal + BFS by lane 0
// ---------------------------------------------------------------------------
__global__ void mst_k(const float* __restrict__ wrow, const float* __restrict__ wcol,
                      int H, int W, int E, int P,
                      float* keys, int* eidx, int* uf, int* adjn, int* deg,
                      int* parent, int* order)
{
    const int b  = blockIdx.x;
    const int tid = threadIdx.x;
    const int T  = blockDim.x;
    const int N  = H * W;
    const int nh = H * (W - 1);
    const int nv = (H - 1) * W;
    float* kb = keys + (size_t)b * P;
    int* ib   = eidx + (size_t)b * P;
    int* ufb  = uf + (size_t)b * N;
    int* adjb = adjn + (size_t)b * N * 4;
    int* degb = deg + (size_t)b * N;
    int* parb = parent + (size_t)b * N;
    int* ordb = order + (size_t)b * N;

    for (int i = tid; i < P; i += T) {
        if (i < E) {
            kb[i] = (i < nh) ? wrow[(size_t)b * nh + i] : wcol[(size_t)b * nv + (i - nh)];
            ib[i] = i;
        } else { kb[i] = 3.4e38f; ib[i] = 0x7fffffff; }
        if (i < N) { ufb[i] = i; degb[i] = 0; }
    }
    __syncthreads();

    for (int k = 2; k <= P; k <<= 1) {
        for (int j = k >> 1; j > 0; j >>= 1) {
            for (int i = tid; i < P; i += T) {
                int l = i ^ j;
                if (l > i) {
                    bool up = ((i & k) == 0);
                    float ka = kb[i], kl = kb[l];
                    int ia = ib[i], il = ib[l];
                    bool gt = (ka > kl) || (ka == kl && ia > il);
                    if (gt == up) { kb[i] = kl; kb[l] = ka; ib[i] = il; ib[l] = ia; }
                }
            }
            __syncthreads();
        }
    }

    if (tid == 0) {
        int cnt = 0;
        for (int s = 0; s < E && cnt < N - 1; ++s) {
            int e = ib[s];
            int src, dst;
            if (e < nh) { int row = e / (W - 1), col = e % (W - 1); src = row * W + col; dst = src + 1; }
            else        { src = e - nh; dst = src + W; }
            int a = src; while (ufb[a] != a) { ufb[a] = ufb[ufb[a]]; a = ufb[a]; }
            int c = dst; while (ufb[c] != c) { ufb[c] = ufb[ufb[c]]; c = ufb[c]; }
            if (a != c) {
                ufb[a] = c;
                adjb[src * 4 + degb[src]++] = dst;
                adjb[dst * 4 + degb[dst]++] = src;
                cnt++;
            }
        }
        for (int n = 0; n < N; ++n) ufb[n] = 0;   // reuse as visited
        parb[0] = 0; ordb[0] = 0; ufb[0] = 1;
        int head = 0, tail = 1;
        while (head < tail) {
            int u = ordb[head++];
            int d = degb[u];
            for (int t = 0; t < d; ++t) {
                int v = adjb[u * 4 + t];
                if (!ufb[v]) { ufb[v] = 1; parb[v] = u; ordb[tail++] = v; }
            }
        }
    }
}

// ---------------------------------------------------------------------------
// Tree filter: per-node edge weight, init, serial scans, fusion add
// ---------------------------------------------------------------------------
__global__ void tree_weights_k(const float* __restrict__ embed,
                               const int* __restrict__ parent,
                               float* __restrict__ w, int B, int N)
{
    int i = blockIdx.x * blockDim.x + threadIdx.x;
    if (i >= B * 16 * N) return;
    int n = i % N;
    int g = (i / N) % 16;
    int b = i / (16 * N);
    int par = parent[(size_t)b * N + n];
    const float* e = embed + (size_t)b * 64 * N;
    float s = 0.0f;
    for (int c = 0; c < 4; ++c) {
        float d = e[(size_t)(4 * g + c) * N + n] - e[(size_t)(4 * g + c) * N + par];
        s += d * d;
    }
    w[i] = (n == 0) ? 0.0f : __expf(-s);
}

__global__ void tree_init_k(const float* __restrict__ feat, float* __restrict__ A,
                            int B, int N)
{
    int i = blockIdx.x * blockDim.x + threadIdx.x;   // over B*16*9*N
    if (i >= B * 16 * 9 * N) return;
    int n  = i % N;
    int ch = (i / N) % 9;
    int g  = (i / (9 * N)) % 16;
    int b  = i / (16 * 9 * N);
    A[i] = (ch < 8) ? feat[(size_t)b * 128 * N + (size_t)(g * 8 + ch) * N + n] : 1.0f;
}

__global__ __launch_bounds__(32) void tree_scan_k(
    const float* __restrict__ w, const int* __restrict__ parent,
    const int* __restrict__ order, float* A, float* Y, int B, int N)
{
    int bg = blockIdx.x;
    int b = bg / 16;
    int ch = threadIdx.x;
    const int* par = parent + (size_t)b * N;
    const int* ord = order + (size_t)b * N;
    const float* wg = w + (size_t)bg * N;
    float* Ac = A + (size_t)bg * 9 * N + (size_t)ch * N;
    float* Yc = Y + (size_t)bg * 9 * N + (size_t)ch * N;
    if (ch < 9) {
        for (int i = N - 1; i >= 1; --i) {           // leaf -> root
            int node = ord[i]; int p = par[node]; float wn = wg[node];
            Ac[p] += wn * Ac[node];
        }
        for (int i = 0; i < N; ++i) {                // root -> leaf
            int node = ord[i]; int p = par[node]; float wn = wg[node];
            float yp = (i == 0) ? 0.0f : Yc[p];
            Yc[node] = (1.0f - wn * wn) * Ac[node] + wn * yp;
        }
    }
}

__global__ void fusion_k(const float* __restrict__ Y, float* __restrict__ lat,
                         int B, int N)
{
    int i = blockIdx.x * blockDim.x + threadIdx.x;   // over B*128*N
    if (i >= B * 128 * N) return;
    int n = i % N;
    int c = (i / N) % 128;
    int b = i / (128 * N);
    int g = c / 8, ch = c % 8;
    const float* Yg = Y + (size_t)(b * 16 + g) * 9 * N;
    float num = Yg[(size_t)ch * N + n];
    float den = Yg[(size_t)8 * N + n];
    lat[i] += num / den;
}

// ---------------------------------------------------------------------------
// Misc elementwise kernels
// ---------------------------------------------------------------------------
__global__ void concat_k(const float* __restrict__ fus, const float* __restrict__ disp,
                         float* __restrict__ out, int B, int N)
{
    int i = blockIdx.x * blockDim.x + threadIdx.x;   // over B*129*N
    if (i >= B * 129 * N) return;
    int n = i % N;
    int c = (i / N) % 129;
    int b = i / (129 * N);
    out[i] = (c < 128) ? fus[(size_t)b * 128 * N + (size_t)c * N + n]
                       : disp[(size_t)b * N + n];
}

__global__ void up2_k(const float* __restrict__ in, float* __restrict__ out,
                      int B, int C, int H, int W)
{
    int H2 = 2 * H, W2 = 2 * W;
    int i = blockIdx.x * blockDim.x + threadIdx.x;
    if (i >= B * C * H2 * W2) return;
    int ox = i % W2;
    int oy = (i / W2) % H2;
    int c  = (i / (W2 * H2)) % C;
    int b  = i / (W2 * H2 * C);
    float sy = oy * (float)(H - 1) / (float)(H2 - 1);
    float sx = ox * (float)(W - 1) / (float)(W2 - 1);
    int y0 = (int)floorf(sy); int y1 = min(y0 + 1, H - 1); float ty = sy - y0;
    int x0 = (int)floorf(sx); int x1 = min(x0 + 1, W - 1); float tx = sx - x0;
    const float* src = in + ((size_t)b * C + c) * H * W;
    float v = (1.0f - ty) * ((1.0f - tx) * src[y0 * W + x0] + tx * src[y0 * W + x1])
            +         ty  * ((1.0f - tx) * src[y1 * W + x0] + tx * src[y1 * W + x1]);
    out[i] = v;
}

__global__ void bnrelu_k(float* __restrict__ y, const float* __restrict__ g,
                         const float* __restrict__ bb, const float* __restrict__ m,
                         const float* __restrict__ v, int B, int HW)
{
    int i = blockIdx.x * blockDim.x + threadIdx.x;
    if (i >= B * 128 * HW) return;
    int c = (i / HW) % 128;
    float x = (y[i] - m[c]) * rsqrtf(v[c] + 1e-5f) * g[c] + bb[c];
    y[i] = fmaxf(x, 0.0f);
}

__global__ void head_out_k(const float* __restrict__ y, const float* __restrict__ w2,
                           const float* __restrict__ b2, float* __restrict__ disp,
                           int B, int HW, int scale)
{
    int i = blockIdx.x * blockDim.x + threadIdx.x;
    if (i >= B * HW) return;
    int b = i / HW, p = i % HW;
    const float* yb = y + (size_t)b * 128 * HW + p;
    float s = b2[0];
    for (int c = 0; c < 128; ++c) s += yb[(size_t)c * HW] * w2[c];
    float sig = 1.0f / (1.0f + __expf(-s));
    disp[i] = scale ? (0.01f + (10.0f - 0.01f) * sig) : sig;
}

// ---------------------------------------------------------------------------
// Host orchestration
// ---------------------------------------------------------------------------
extern "C" void kernel_launch(void* const* d_in, const int* in_sizes, int n_in,
                              void* d_out, int out_size, void* d_ws, size_t ws_size,
                              hipStream_t stream)
{
    const int B = 4;
    // pytree (sorted-dict) flattening: fm1..fm4, embed[0..2],
    // head[h]{b,b2,g,m,v,w1,w2}, latent[0..3], refine[0..2]
    const float* fms[4] = { (const float*)d_in[3], (const float*)d_in[2],
                            (const float*)d_in[1], (const float*)d_in[0] }; // fm4,fm3,fm2,fm1
    const int fmC[4] = { 2048, 1024, 512, 256 };
    const float* embedW[3] = { (const float*)d_in[4], (const float*)d_in[5], (const float*)d_in[6] };
    struct HeadP { const float *bb, *b2, *g, *m, *v, *w1, *w2; };
    HeadP hp[3];
    for (int h = 0; h < 3; ++h) {
        int base = 7 + 7 * h;
        hp[h].bb = (const float*)d_in[base + 0];
        hp[h].b2 = (const float*)d_in[base + 1];
        hp[h].g  = (const float*)d_in[base + 2];
        hp[h].m  = (const float*)d_in[base + 3];
        hp[h].v  = (const float*)d_in[base + 4];
        hp[h].w1 = (const float*)d_in[base + 5];
        hp[h].w2 = (const float*)d_in[base + 6];
    }
    const float* latW[4] = { (const float*)d_in[28], (const float*)d_in[29],
                             (const float*)d_in[30], (const float*)d_in[31] };
    const float* refW[3] = { (const float*)d_in[32], (const float*)d_in[33],
                             (const float*)d_in[34] };

    // workspace carving
    char* ws = (char*)d_ws;
    size_t off = 0;
    auto alloc = [&](size_t bytes) -> void* {
        void* p = (void*)(ws + off);
        off += (bytes + 255) & ~(size_t)255;
        return p;
    };
    const int Nmax = 32 * 104;        // 3328
    const int N2max = 64 * 208;       // 13312
    float* latbuf  = (float*)alloc((size_t)B * 128 * Nmax * 4);
    float* lastfm  = (float*)alloc((size_t)B * 128 * N2max * 4);
    float* refined = (float*)alloc((size_t)B * 128 * Nmax * 4);
    float* embbuf  = (float*)alloc((size_t)B * 64 * Nmax * 4);
    float* Abuf    = (float*)alloc((size_t)B * 16 * 9 * Nmax * 4);
    float* Ybuf    = (float*)alloc((size_t)B * 16 * 9 * Nmax * 4);
    float* wbuf    = (float*)alloc((size_t)B * 16 * Nmax * 4);
    float* wrow    = (float*)alloc((size_t)B * 32 * 103 * 4);
    float* wcol    = (float*)alloc((size_t)B * 31 * 104 * 4);
    float* keys    = (float*)alloc((size_t)B * 8192 * 4);
    int*   eidx    = (int*)alloc((size_t)B * 8192 * 4);
    int*   uf      = (int*)alloc((size_t)B * Nmax * 4);
    int*   adjn    = (int*)alloc((size_t)B * Nmax * 4 * 4);
    int*   deg     = (int*)alloc((size_t)B * Nmax * 4);
    int*   parent  = (int*)alloc((size_t)B * Nmax * 4);
    int*   order   = (int*)alloc((size_t)B * Nmax * 4);
    float* inp129  = (float*)alloc((size_t)B * 129 * Nmax * 4);
    float* headY   = (float*)alloc((size_t)B * 128 * N2max * 4);
    float* disp1   = (float*)alloc((size_t)B * 16 * 52 * 4);
    float* disp2   = (float*)alloc((size_t)B * 32 * 104 * 4);

    auto conv3 = [&](const float* in, const float* w, float* out,
                     int Cin, int H, int W, int Cout) {
        int HW = H * W;
        dim3 g((HW + 63) / 64, (Cout + 15) / 16, B);
        conv_wmma_t<3, 1><<<g, dim3(128), 0, stream>>>(in, w, out, B, Cin, H, W, Cout);
    };
    auto conv1 = [&](const float* in, const float* w, float* out,
                     int Cin, int H, int W, int Cout) {
        int HW = H * W;
        dim3 g((HW + 63) / 64, (Cout + 15) / 16, B);
        conv_wmma_t<1, 0><<<g, dim3(128), 0, stream>>>(in, w, out, B, Cin, H, W, Cout);
    };
    auto ew = [&](int tot) { return dim3((tot + 255) / 256); };
    auto p2 = [](int x) { int p = 1; while (p < x) p <<= 1; return p; };

    const int Hs[4] = { 4, 8, 16, 32 };
    const int Ws[4] = { 13, 26, 52, 104 };

    // idx = 0: latent conv on fm4, refined = latent, lastfm = up2(refined)
    conv3(fms[0], latW[0], refined, fmC[0], Hs[0], Ws[0], 128);
    up2_k<<<ew(B * 128 * 4 * Hs[0] * Ws[0]), 256, 0, stream>>>(refined, lastfm, B, 128, Hs[0], Ws[0]);

    const float* dispPrev = nullptr;
    for (int idx = 1; idx < 4; ++idx) {
        int H = Hs[idx], W = Ws[idx], N = H * W;
        // latent conv
        conv3(fms[idx], latW[idx], latbuf, fmC[idx], H, W, 128);
        // MST on latent edge distances
        int nh = H * (W - 1), nv = (H - 1) * W, E = nh + nv;
        edge_dists_k<<<ew(B * E), 256, 0, stream>>>(latbuf, wrow, wcol, B, H, W);
        int P = p2(E);
        mst_k<<<B, 256, 0, stream>>>(wrow, wcol, H, W, E, P,
                                     keys, eidx, uf, adjn, deg, parent, order);
        // embed conv (1x1) on lastfm
        conv1(lastfm, embedW[idx - 1], embbuf, 128, H, W, 64);
        // tree filter
        tree_weights_k<<<ew(B * 16 * N), 256, 0, stream>>>(embbuf, parent, wbuf, B, N);
        tree_init_k<<<ew(B * 16 * 9 * N), 256, 0, stream>>>(lastfm, Abuf, B, N);
        tree_scan_k<<<B * 16, 32, 0, stream>>>(wbuf, parent, order, Abuf, Ybuf, B, N);
        fusion_k<<<ew(B * 128 * N), 256, 0, stream>>>(Ybuf, latbuf, B, N);
        // refine conv (concat previous disp beyond idx==1)
        const float* rin = latbuf; int cin = 128;
        if (idx >= 2) {
            concat_k<<<ew(B * 129 * N), 256, 0, stream>>>(latbuf, dispPrev, inp129, B, N);
            rin = inp129; cin = 129;
        }
        conv1(rin, refW[idx - 1], refined, cin, H, W, 128);
        // upsample
        up2_k<<<ew(B * 128 * 4 * N), 256, 0, stream>>>(refined, lastfm, B, 128, H, W);
        int H2 = 2 * H, W2 = 2 * W, HW2 = H2 * W2;
        // predict head
        conv3(lastfm, hp[idx - 1].w1, headY, 128, H2, W2, 128);
        bnrelu_k<<<ew(B * 128 * HW2), 256, 0, stream>>>(headY, hp[idx - 1].g, hp[idx - 1].bb,
                                                        hp[idx - 1].m, hp[idx - 1].v, B, HW2);
        float* dst = (idx == 1) ? disp1 : (idx == 2) ? disp2 : (float*)d_out;
        head_out_k<<<ew(B * HW2), 256, 0, stream>>>(headY, hp[idx - 1].w2, hp[idx - 1].b2,
                                                    dst, B, HW2, idx == 3 ? 1 : 0);
        dispPrev = dst;
    }
}
